// CodebookSingleW_66331474919622
// MI455X (gfx1250) — matrix operand order repaired and checked by reading, not compiled, not run on hardware
//
#include <hip/hip_runtime.h>

#define IN_DIM 4096
#define HDIM   4096
#define BATCH  8192
#define KCB    256

typedef __attribute__((ext_vector_type(16))) __bf16 v16bf;
typedef __attribute__((ext_vector_type(8)))  float  v8f;

union Frag16 { v16bf v; uint4 u[2]; };
union Pack8  { __bf16 h[8]; uint4 u; };
union Pack4  { __bf16 h[4]; uint2 u; };

__device__ __forceinline__ void async_copy_2xb128(unsigned lds_off, const void* gaddr) {
  // GLOBAL_LOAD_ASYNC_TO_LDS_B128: vdst = LDS byte addr VGPR, vaddr = 64-bit global
  // addr; IOFFSET applies to both LDS and global addresses (ISA 08 §4.4).
  asm volatile("global_load_async_to_lds_b128 %0, %1, off"
               :: "v"(lds_off), "v"(gaddr) : "memory");
  asm volatile("global_load_async_to_lds_b128 %0, %1, off offset:16"
               :: "v"(lds_off), "v"(gaddr) : "memory");
}
__device__ __forceinline__ void wait_async0() {
  asm volatile("s_wait_asynccnt 0x0" ::: "memory");
}

// ---------------------------------------------------------------------------
// Dequantize codebook[indices] -> W (bf16 [IN_DIM][H]) and WT (bf16 [H][IN_DIM])
// ---------------------------------------------------------------------------
__global__ __launch_bounds__(256) void dequant_kernel(
    const int* __restrict__ indices, const float* __restrict__ codebook,
    __bf16* __restrict__ W, __bf16* __restrict__ WT) {
  __shared__ float cb[KCB];
  __shared__ __bf16 tile[32][34];
  const int t = threadIdx.x;
  if (t < KCB) cb[t] = codebook[t];
  __syncthreads();

  const int i0 = blockIdx.y * 32;   // in_dim
  const int j0 = blockIdx.x * 32;   // H
  const int lr = t >> 3;            // 0..31
  const int lc = (t & 7) * 4;       // 0,4,...,28

  const int4 idx4 = *(const int4*)&indices[(long)(i0 + lr) * HDIM + j0 + lc];
  Pack4 w;
  w.h[0] = (__bf16)cb[idx4.x];
  w.h[1] = (__bf16)cb[idx4.y];
  w.h[2] = (__bf16)cb[idx4.z];
  w.h[3] = (__bf16)cb[idx4.w];

  *(uint2*)&W[(long)(i0 + lr) * HDIM + j0 + lc] = w.u;
  tile[lr][lc + 0] = w.h[0]; tile[lr][lc + 1] = w.h[1];
  tile[lr][lc + 2] = w.h[2]; tile[lr][lc + 3] = w.h[3];
  __syncthreads();

  Pack4 o;
  o.h[0] = tile[lc + 0][lr]; o.h[1] = tile[lc + 1][lr];
  o.h[2] = tile[lc + 2][lr]; o.h[3] = tile[lc + 3][lr];
  *(uint2*)&WT[(long)(j0 + lr) * IN_DIM + i0 + lc] = o.u;
}

// ---------------------------------------------------------------------------
// x (f32) -> bf16, 8 elements per thread
// ---------------------------------------------------------------------------
__global__ __launch_bounds__(256) void xcvt_kernel(
    const float* __restrict__ x, __bf16* __restrict__ xb) {
  const long i = ((long)blockIdx.x * 256 + threadIdx.x) * 8;
  const float4 a = *(const float4*)(x + i);
  const float4 b = *(const float4*)(x + i + 4);
  Pack8 o;
  o.h[0] = (__bf16)a.x; o.h[1] = (__bf16)a.y;
  o.h[2] = (__bf16)a.z; o.h[3] = (__bf16)a.w;
  o.h[4] = (__bf16)b.x; o.h[5] = (__bf16)b.y;
  o.h[6] = (__bf16)b.z; o.h[7] = (__bf16)b.w;
  *(uint4*)(xb + i) = o.u;
}

// ---------------------------------------------------------------------------
// Tiled bf16 WMMA GEMM: Out[M][N] = A[M][K] * Bmat[N][K]^T (+bias, +C19 act).
// Block tile 128x128, BK=32, 8 waves (2x4), wave tile 64x32 (4x2 WMMA tiles).
// Double-buffered LDS fed by async global->LDS; unrolled x2 so buffer index
// is static; last two K-steps peeled.
// ---------------------------------------------------------------------------
template <bool FIRST>
__global__ __launch_bounds__(256) void gemm_kernel(
    const __bf16* __restrict__ Amat,   // [M][K]
    const __bf16* __restrict__ Bmat,   // [N][K]
    const float* __restrict__ bias,
    const float* __restrict__ craw_arr,
    const float* __restrict__ rhoraw_arr,
    void* __restrict__ Outptr,
    int M, int N, int K) {
  constexpr int BM = 128, BN = 128, BK = 32, KP = 40;  // 80B row stride, 16B aligned
  __shared__ unsigned short As[2][BM][KP];
  __shared__ unsigned short Bs[2][BN][KP];

  const int t = threadIdx.x;
  const int wave = t >> 5, lane = t & 31;
  const int blockM = blockIdx.y * BM, blockN = blockIdx.x * BN;
  const int waveM = (wave >> 2) * 64;   // {0,64}
  const int waveN = (wave & 3) * 32;    // {0,32,64,96}

  // staging: thread -> (row, 16-element k segment); 32B per thread per tile
  const int srow = t >> 1;
  const int sseg = (t & 1) * 16;

  const __bf16* pa = Amat + (long)(blockM + srow) * K + sseg;
  const __bf16* pb = Bmat + (long)(blockN + srow) * K + sseg;
  unsigned ldsA[2], ldsB[2];
  ldsA[0] = (unsigned)(uintptr_t)&As[0][srow][sseg];
  ldsA[1] = (unsigned)(uintptr_t)&As[1][srow][sseg];
  ldsB[0] = (unsigned)(uintptr_t)&Bs[0][srow][sseg];
  ldsB[1] = (unsigned)(uintptr_t)&Bs[1][srow][sseg];

  const v8f vzero = {0.f, 0.f, 0.f, 0.f, 0.f, 0.f, 0.f, 0.f};
  v8f acc[4][2];
#pragma unroll
  for (int mi = 0; mi < 4; ++mi)
#pragma unroll
    for (int ni = 0; ni < 2; ++ni) acc[mi][ni] = vzero;

  // ISA VGPR layouts (16-bit, wave32):
  //  A 16x32: lanes 0-15 hold K {0..7,16..23}; lanes 16-31 hold K {8..15,24..31}
  //  B 32x16: lanes 0-15 hold K {0..7,16..23}; lanes 16-31 hold K {8..15,24..31}
  const int mrow = lane & 15;
  const int kA = (lane < 16) ? 0 : 8;
  const int kB = (lane < 16) ? 0 : 8;

  auto stage = [&](int buf) {
    async_copy_2xb128(ldsA[buf], pa);
    async_copy_2xb128(ldsB[buf], pb);
  };
  auto compute = [&](int buf) {
    Frag16 b[2];
#pragma unroll
    for (int ni = 0; ni < 2; ++ni) {
      const unsigned short* p = &Bs[buf][waveN + ni * 16 + mrow][0];
      b[ni].u[0] = *(const uint4*)(p + kB);
      b[ni].u[1] = *(const uint4*)(p + kB + 16);
    }
#pragma unroll
    for (int mi = 0; mi < 4; ++mi) {
      Frag16 a;
      const unsigned short* p = &As[buf][waveM + mi * 16 + mrow][0];
      a.u[0] = *(const uint4*)(p + kA);
      a.u[1] = *(const uint4*)(p + kA + 16);
#pragma unroll
      for (int ni = 0; ni < 2; ++ni)
        acc[mi][ni] = __builtin_amdgcn_wmma_f32_16x16x32_bf16(
            false, a.v, false, b[ni].v, (short)0, acc[mi][ni], false, false);
    }
  };

  const int KT = K / BK;  // 128, even
  stage(0);
  wait_async0();
  __syncthreads();

  for (int kt = 0; kt < KT - 2; kt += 2) {
    pa += BK; pb += BK;
    stage(1);
    compute(0);
    wait_async0();
    __syncthreads();

    pa += BK; pb += BK;
    stage(0);
    compute(1);
    wait_async0();
    __syncthreads();
  }
  pa += BK; pb += BK;
  stage(1);
  compute(0);
  wait_async0();
  __syncthreads();
  compute(1);

  // Epilogue. C/D layout: VGPR r, lane L -> (M = r + (L<16?0:8), N = L&15)
#pragma unroll
  for (int ni = 0; ni < 2; ++ni) {
    const int col = blockN + waveN + ni * 16 + mrow;
    const float bv = bias[col];
    float cc = 0.f, icc = 0.f, rho = 0.f;
    if constexpr (FIRST) {
      const float cr = craw_arr[col];
      const float rr = rhoraw_arr[col];
      cc = __expf(cr);
      icc = __expf(-cr);
      rho = 1.0f / (1.0f + __expf(-rr));
    }
#pragma unroll
    for (int mi = 0; mi < 4; ++mi) {
      const int rbase = blockM + waveM + mi * 16 + ((lane < 16) ? 0 : 8);
#pragma unroll
      for (int r = 0; r < 8; ++r) {
        const float z = acc[mi][ni][r] + bv;
        if constexpr (FIRST) {
          // tanh(y) = 1 - 2/(exp(2y)+1)
          const float y = z * icc;
          const float th = 1.0f - 2.0f * __builtin_amdgcn_rcpf(__expf(2.0f * y) + 1.0f);
          const float f = rho * z + (1.0f - rho) * cc * th;
          ((__bf16*)Outptr)[(long)(rbase + r) * N + col] = (__bf16)f;
        } else {
          ((float*)Outptr)[(long)(rbase + r) * N + col] = z;
        }
      }
    }
  }
}

extern "C" void kernel_launch(void* const* d_in, const int* in_sizes, int n_in,
                              void* d_out, int out_size, void* d_ws, size_t ws_size,
                              hipStream_t stream) {
  const float* x        = (const float*)d_in[0];
  const float* codebook = (const float*)d_in[1];
  const int*   indices  = (const int*)d_in[2];
  const float* b1       = (const float*)d_in[3];
  const float* b2       = (const float*)d_in[4];
  const float* c19_c    = (const float*)d_in[5];
  const float* c19_rho  = (const float*)d_in[6];

  __bf16* W  = (__bf16*)d_ws;                          // [IN_DIM][H]
  __bf16* WT = W  + (size_t)IN_DIM * HDIM;             // [H][IN_DIM]
  __bf16* h  = WT + (size_t)IN_DIM * HDIM;             // [B][H]
  __bf16* xb = h  + (size_t)BATCH * HDIM;              // [B][IN_DIM]

  dequant_kernel<<<dim3(HDIM / 32, IN_DIM / 32), 256, 0, stream>>>(
      indices, codebook, W, WT);

  xcvt_kernel<<<dim3((BATCH * IN_DIM) / (256 * 8)), 256, 0, stream>>>(x, xb);

  gemm_kernel<true><<<dim3(HDIM / 128, BATCH / 128), 256, 0, stream>>>(
      xb, WT, b1, c19_c, c19_rho, h, BATCH, HDIM, IN_DIM);

  gemm_kernel<false><<<dim3(IN_DIM / 128, BATCH / 128), 256, 0, stream>>>(
      h, W, b2, nullptr, nullptr, d_out, BATCH, IN_DIM, HDIM);
}
